// SecondaryCapsule_6854767805119
// MI455X (gfx1250) — matrix operand order repaired and checked by reading, not compiled, not run on hardware
//
#include <hip/hip_runtime.h>
#include <hip/hip_bf16.h>

// Capsule dynamic routing for MI455X (gfx1250, wave32, WMMA).
// u_hat (604MB) is never materialized: every routing pass is a GEMM over the
// L2-resident x (9.4MB) and W (18.9MB) using v_wmma_f32_16x16x32_f16.
// Operands are staged through LDS (async global->LDS where available) so the
// WMMA pipe is fed by ds_read_b128 instead of per-lane scattered VMEM.

typedef __attribute__((ext_vector_type(16))) _Float16 v16h;
typedef __attribute__((ext_vector_type(8)))  float    v8f;
typedef int i32x4_t __attribute__((vector_size(16)));

#define B_   256
#define R_   1152
#define K_   32
#define O_   16
#define C_   8
#define RC_  (R_ * C_)      // 9216  (GEMM-S contraction dim)
#define KO_  (K_ * O_)      // 512   (N dim)
#define RK_  (R_ * K_)      // 36864 (routing state size)
#define BKO_ (B_ * K_ * O_) // 131072 (s / v / output size)

#if __has_builtin(__builtin_amdgcn_global_load_async_to_lds_b128) && \
    __has_builtin(__builtin_amdgcn_s_wait_asynccnt)
#define HAVE_ASYNC_LDS 1
#else
#define HAVE_ASYNC_LDS 0
#endif

// Copy 16 bytes global -> LDS. Async path uses the gfx1250 ASYNCcnt engine.
// Builtin prototype (per hipcc diagnostic): (int4 AS1*, int4 AS3*, imm, imm).
__device__ __forceinline__ void copy16_g2l(const float* __restrict__ g,
                                           float* __restrict__ l)
{
#if HAVE_ASYNC_LDS
    __builtin_amdgcn_global_load_async_to_lds_b128(
        (__attribute__((address_space(1))) i32x4_t*)(i32x4_t*)const_cast<float*>(g),
        (__attribute__((address_space(3))) i32x4_t*)(i32x4_t*)l,
        0, 0);
#else
    *(float4*)l = *(const float4*)g;
#endif
}
__device__ __forceinline__ void wait_g2l()
{
#if HAVE_ASYNC_LDS
    __builtin_amdgcn_s_wait_asynccnt(0);
#endif
}

// A-matrix (16x32, f16) K index for fragment half h (0..15), lane group g:
// VGPR v=h/2: lanes0-15 K = {0..7,16..23}, lanes16-31 +8 (ISA 7.12.2 table).
__device__ __forceinline__ int kidxA(int h, int g) {
    return (h & 7) + ((h >> 3) << 4) + (g << 3);
}
// B k index, contiguous order (not the ISA table): lanes0-15 K=0..15, lanes16-31 K=16..31.
__device__ __forceinline__ int kidxB(int h, int g) {
    return (g << 4) + h;
}

// -------- GEMM-S: s[b, k*16+o] += sum_rc x[b,rc] * c[r,k] * W[r,k,o,c] -----
// M=256 (b), N=512 (ko), Kdim=9216 (rc). 4 waves/block share one k column:
// the c-scaled W tile is staged once per K-step into LDS as f16 and each
// lane's whole B fragment is one aligned 32B LDS read. 18-way K split with
// f32 atomic accumulation into s.
__global__ void __launch_bounds__(128) gemm_s_kernel(
    const float* __restrict__ x, const float* __restrict__ W,
    const float* __restrict__ cij, float* __restrict__ s)
{
    const int t    = threadIdx.x;
    const int lane = t & 31;
    const int wave = t >> 5;
    const int g    = (lane >> 4) & 1;
    const int l16  = lane & 15;

    const int kb    = blockIdx.x >> 2;                    // capsule column (uniform)
    const int bBase = (((blockIdx.x & 3) << 2) | wave) * 16;
    const int rc0   = blockIdx.y * 512;                   // K-chunk base (18 chunks)

    // B tile in LDS: bt[o][K], row padded 32->40 halves so the per-lane 32B
    // fragment read (byte offset l16*80 + g*32) is 16B aligned and the 16
    // lanes of a group hit distinct banks (20*l16 mod 64 all distinct).
    __shared__ __align__(16) _Float16 bt[16 * 40];

    // staging role of this thread: chunk j (one r), 4 consecutive floats
    const int j    = t >> 5;          // 0..3   (r offset inside 32-rc step)
    const int idx  = (t & 31) * 4;    // 0..124 (o*8 + c)
    const int o_st = idx >> 3;
    const int c_st = idx & 7;

    v8f acc = {};
    const float* xrow = x + (bBase + l16) * RC_;

    for (int st = 0; st < 16; ++st) {
        const int rcBase = rc0 + st * 32;   // 32 rc = 4 r's x 8 c
        const int rbase  = rcBase >> 3;

        __syncthreads();  // previous step's fragment reads are done
        {
            const float  cj = cij[(rbase + j) * K_ + kb];
            const float4 wf = *(const float4*)&W[(rbase + j) * 4096 + kb * 128 + idx];
            _Float16* dst = &bt[o_st * 40 + j * 8 + c_st];
            dst[0] = (_Float16)(wf.x * cj);
            dst[1] = (_Float16)(wf.y * cj);
            dst[2] = (_Float16)(wf.z * cj);
            dst[3] = (_Float16)(wf.w * cj);
            if (st < 15)  // pull next W chunk toward the caches
                __builtin_prefetch(&W[(rbase + 4 + j) * 4096 + kb * 128 + idx], 0, 1);
        }
        __syncthreads();

        // A fragment: one contiguous 16-float span per lane -> 4x b128 loads.
        const float* ap = xrow + rcBase;
        const float4 a0 = *(const float4*)&ap[g * 16];
        const float4 a1 = *(const float4*)&ap[g * 16 + 4];
        const float4 a2 = *(const float4*)&ap[g * 16 + 8];
        const float4 a3 = *(const float4*)&ap[g * 16 + 12];
        v16h af;
        af[0]  = (_Float16)a0.x; af[1]  = (_Float16)a0.y;
        af[2]  = (_Float16)a0.z; af[3]  = (_Float16)a0.w;
        af[4]  = (_Float16)a1.x; af[5]  = (_Float16)a1.y;
        af[6]  = (_Float16)a1.z; af[7]  = (_Float16)a1.w;
        af[8]  = (_Float16)a2.x; af[9]  = (_Float16)a2.y;
        af[10] = (_Float16)a2.z; af[11] = (_Float16)a2.w;
        af[12] = (_Float16)a3.x; af[13] = (_Float16)a3.y;
        af[14] = (_Float16)a3.z; af[15] = (_Float16)a3.w;

        // B fragment: halves K = g*16..g*16+15 for col o=l16 -> one 32B read.
        const v16h bf = *(const v16h*)&bt[l16 * 40 + (g << 4)];

        acc = __builtin_amdgcn_wmma_f32_16x16x32_f16(false, af, false, bf,
                                                     (short)0, acc, false, false);
    }

    // C/D layout: VGPR i, lane group g -> row M = i + 8g, col N = lane&15.
    #pragma unroll
    for (int i = 0; i < 8; ++i) {
        const int brow = bBase + i + 8 * g;
        atomicAdd(&s[brow * KO_ + kb * 16 + l16], acc[i]);
    }
}

// -------- GEMM-A: a[r,k] = (1/B) sum_{o,c} W[r,k,o,c] * P[rc,ko] -----------
// P = x^T v : M=9216 (rc), N=512 (ko), Kdim=256 (b). x-transpose and v tiles
// are staged to LDS with async global->LDS copies; the epilogue fuses the
// W-weighted reduction of each 16x16 P tile (2 r's x 1 k) into a[r,k].
__global__ void __launch_bounds__(128) gemm_a_kernel(
    const float* __restrict__ x, const float* __restrict__ v,
    const float* __restrict__ W, float* __restrict__ aij)
{
    const int t    = threadIdx.x;
    const int lane = t & 31;
    const int wave = t >> 5;
    const int g    = (lane >> 4) & 1;
    const int l16  = lane & 15;

    const int tid4   = blockIdx.x * 4;      // 4 consecutive M tiles, same k
    const int kb     = tid4 / 576;
    const int tileM0 = tid4 - kb * 576;
    const int rcB0   = tileM0 * 16;         // block covers rc span of 64
    const int rcBase = rcB0 + wave * 16;    // this wave's M tile
    const int koBase = kb * 16;

    // tileA[b][rc_local]: row padded 64->68 floats (272B, 16B aligned) so the
    // two lane groups (b rows differing by 8) land on distinct bank sets.
    __shared__ __align__(16) float tileA[32 * 68];
    __shared__ __align__(16) float tileV[32 * 16];

    const int row = t >> 2;   // 0..31 : b row inside the step
    const int seg = t & 3;    // 16-float segment (tileA) / float4 col (tileV)

    v8f acc = {};

    for (int bBase = 0; bBase < B_; bBase += 32) {
        __syncthreads();  // previous step's fragment reads are done

        // stage x block: 32 b-rows x 64 rc  (each thread copies 4x 16B)
        const float* xsrc = x + (bBase + row) * RC_ + rcB0 + seg * 16;
        float*       xdst = &tileA[row * 68 + seg * 16];
        #pragma unroll
        for (int q = 0; q < 4; ++q)
            copy16_g2l(xsrc + q * 4, xdst + q * 4);

        // stage v block: 32 b-rows x 16 o  (each thread copies 1x 16B)
        copy16_g2l(v + (bBase + row) * KO_ + koBase + seg * 4,
                   &tileV[row * 16 + seg * 4]);

        wait_g2l();
        __syncthreads();

        // A fragment: A[m=rc][kdim=b] = x^T, read down the b rows of tileA.
        const int rcL = wave * 16 + l16;
        v16h af, bf;
        #pragma unroll
        for (int h = 0; h < 16; ++h) {
            const int kk = kidxA(h, g);
            af[h] = (_Float16)tileA[kk * 68 + rcL];
        }
        #pragma unroll
        for (int h = 0; h < 16; ++h) {
            const int kk = kidxA(h, g);
            bf[h] = (_Float16)tileV[kk * 16 + l16];
        }
        acc = __builtin_amdgcn_wmma_f32_16x16x32_f16(false, af, false, bf,
                                                     (short)0, acc, false, false);
    }

    // Tile rows rcBase..+15 = r0 (g=0) and r0+1 (g=1); c = VGPR index i;
    // cols = the 16 o's of capsule kb, o = lane&15.
    const int r = (rcBase >> 3) + g;
    const float* wrow = W + r * 4096 + kb * 128 + l16 * 8;
    float p = 0.f;
    #pragma unroll
    for (int i = 0; i < 8; ++i) p += acc[i] * wrow[i];
    // butterfly sum across the 16 lanes sharing this r (masks <16 stay in group)
    p += __shfl_xor(p, 1);
    p += __shfl_xor(p, 2);
    p += __shfl_xor(p, 4);
    p += __shfl_xor(p, 8);
    if (l16 == 0) aij[r * K_ + kb] = p * (1.0f / (float)B_);  // mean over batch
}

// -------- b += a ; c = softmax_R(b) (one block per k) -----------------------
__global__ void __launch_bounds__(256) update_softmax_kernel(
    float* __restrict__ bstate, const float* __restrict__ aij,
    float* __restrict__ cij)
{
    const int k   = blockIdx.x;
    const int tid = threadIdx.x;
    __shared__ float red[256];

    float vals[5];
    int cnt = 0;
    float lmax = -3.0e38f;
    for (int r = tid; r < R_; r += 256) {
        float bv = bstate[r * K_ + k] + aij[r * K_ + k];
        bstate[r * K_ + k] = bv;
        vals[cnt++] = bv;
        lmax = fmaxf(lmax, bv);
    }
    red[tid] = lmax;
    __syncthreads();
    for (int st = 128; st > 0; st >>= 1) {
        if (tid < st) red[tid] = fmaxf(red[tid], red[tid + st]);
        __syncthreads();
    }
    const float mx = red[0];
    __syncthreads();
    float lsum = 0.f;
    for (int i = 0; i < cnt; ++i) lsum += __expf(vals[i] - mx);
    red[tid] = lsum;
    __syncthreads();
    for (int st = 128; st > 0; st >>= 1) {
        if (tid < st) red[tid] += red[tid + st];
        __syncthreads();
    }
    const float inv = 1.0f / red[0];
    cnt = 0;
    for (int r = tid; r < R_; r += 256)
        cij[r * K_ + k] = __expf(vals[cnt++] - mx) * inv;
}

// -------- squash along O ----------------------------------------------------
__global__ void squash_kernel(const float* __restrict__ s, float* __restrict__ out)
{
    const int t = blockIdx.x * blockDim.x + threadIdx.x;
    if (t >= B_ * K_) return;
    const float* sp = s + t * O_;
    float tmp[O_];
    float sq = 0.f;
    #pragma unroll
    for (int i = 0; i < O_; ++i) { tmp[i] = sp[i]; sq += tmp[i] * tmp[i]; }
    const float scale = (sq / (1.f + sq)) * rsqrtf(sq + 1e-8f);
    #pragma unroll
    for (int i = 0; i < O_; ++i) out[t * O_ + i] = tmp[i] * scale;
}

__global__ void zero_kernel(float* __restrict__ p, int n)
{
    const int t = blockIdx.x * blockDim.x + threadIdx.x;
    if (t < n) p[t] = 0.f;
}

extern "C" void kernel_launch(void* const* d_in, const int* in_sizes, int n_in,
                              void* d_out, int out_size, void* d_ws, size_t ws_size,
                              hipStream_t stream)
{
    (void)in_sizes; (void)n_in; (void)out_size; (void)ws_size;
    const float* x = (const float*)d_in[0];   // [B, R, C] = [256, 1152, 8]
    const float* W = (const float*)d_in[1];   // [R, K, O, C] = [1152, 32, 16, 8]
    float* out = (float*)d_out;               // [B, K, O, 1] -> 131072 f32
    float* ws  = (float*)d_ws;

    float* bstate = ws;                 // [R,K]
    float* cij    = bstate + RK_;       // [R,K]
    float* s      = cij + RK_;          // [B,K,O]
    float* vbuf   = s + BKO_;           // [B,K,O]
    float* aij    = vbuf + BKO_;        // [R,K]   (total ~1.5 MB)

    auto zg = [](int n) { return dim3((unsigned)((n + 255) / 256)); };

    // b = 0, a = 0 -> c0 = softmax(0) = 1/R (uniform), matching the reference.
    zero_kernel<<<zg(RK_), 256, 0, stream>>>(bstate, RK_);
    zero_kernel<<<zg(RK_), 256, 0, stream>>>(aij, RK_);
    update_softmax_kernel<<<K_, 256, 0, stream>>>(bstate, aij, cij);

    for (int it = 0; it < 3; ++it) {
        zero_kernel<<<zg(BKO_), 256, 0, stream>>>(s, BKO_);
        gemm_s_kernel<<<dim3(128, 18), 128, 0, stream>>>(x, W, cij, s);
        squash_kernel<<<zg(B_ * K_), 256, 0, stream>>>(s, it == 2 ? out : vbuf);
        if (it < 2) {
            gemm_a_kernel<<<dim3(4608), 128, 0, stream>>>(x, vbuf, W, aij);
            update_softmax_kernel<<<K_, 256, 0, stream>>>(bstate, aij, cij);
        }
    }
}
